// EGNN_Module_68195490726194
// MI455X (gfx1250) — matrix operand order repaired and checked by reading, not compiled, not run on hardware
//
#include <hip/hip_runtime.h>

#define NND    256
#define HD     64
#define EDIM   4
#define KIN    133
#define KP     160
#define BATCH  8
#define LAYERS 4
#define EPSV   1e-8f
#define REGV   0.01f

typedef __attribute__((ext_vector_type(16))) _Float16 v16h;
typedef __attribute__((ext_vector_type(8)))  _Float16 v8h;
typedef __attribute__((ext_vector_type(2)))  _Float16 v2h;
typedef __attribute__((ext_vector_type(8)))  float    v8f;

// silu via hardware v_rcp_f32 (results feed f16 WMMA fragments; IEEE divide's
// div_scale/div_fixup sequence (~8 VALU) is wasted precision here).
__device__ __forceinline__ float silu_f(float v) {
  return v * __builtin_amdgcn_rcpf(1.0f + __expf(-v));
}

__device__ __forceinline__ v8f wmma16(v16h a, v16h b, v8f c) {
  // D = A(16x32 f16) * B(32x16 f16) + C(16x16 f32)
  return __builtin_amdgcn_wmma_f32_16x16x32_f16(false, a, false, b, (short)0, c, false, false);
}

// A fragment (16x32 f16 tile) from row-major LDS [*, ld]; two 16B ds_load_b128 per lane.
// ISA 7.12.2: lanes 0-15 -> M=lane, K base 0; lanes 16-31 -> M=lane-16, K base 8;
// halves 0..7 = K in [kb, kb+8), halves 8..15 = K in [16+kb, 24+kb).
__device__ __forceinline__ v16h frag_a(const _Float16* s, int ld, int row0, int k0, int lane) {
  const int m  = row0 + (lane & 15);
  const int kb = (lane >> 4) << 3;
  const _Float16* p = s + m * ld + k0 + kb;   // 16B aligned for all call sites
  const v8h lo = *(const v8h*)(p);
  const v8h hi = *(const v8h*)(p + 16);
  return __builtin_shufflevector(lo, hi, 0, 1, 2, 3, 4, 5, 6, 7, 8, 9, 10, 11, 12, 13, 14, 15);
}

// B fragment (32x16 tile) from TRANSPOSED weights sT[n][k] (row stride ldk):
// lane needs two runs of 8 consecutive K for fixed N -> two contiguous ds_load_b128.
// lanes 0-15: N=lane, K in [k0, k0+8) and [k0+16, k0+24); lanes 16-31: N=lane-16, K in [k0+8, k0+16) and [k0+24, k0+32).
__device__ __forceinline__ v16h frag_bT(const _Float16* sT, int ldk, int k0, int n0, int lane) {
  const int n  = n0 + (lane & 15);
  const int kb = (lane >> 4) << 3;
  const _Float16* p = sT + n * ldk + k0 + kb;
  const v8h lo = *(const v8h*)(p);
  const v8h hi = *(const v8h*)(p + 16);
  return __builtin_shufflevector(lo, hi, 0, 1, 2, 3, 4, 5, 6, 7, 8, 9, 10, 11, 12, 13, 14, 15);
}

// ---------------------------------------------------------------------------
// embedding_in: h[node,16] -> silu(h@W1+b1)@W2+b2 -> h0[node,64]. Tiny, VALU.
// ---------------------------------------------------------------------------
__global__ __launch_bounds__(64) void embed_in_kernel(
    const float* __restrict__ hin, const float* __restrict__ w1, const float* __restrict__ b1,
    const float* __restrict__ w2, const float* __restrict__ b2, float* __restrict__ hout) {
  __shared__ float hs[16];
  __shared__ float ts[HD];
  const int node = blockIdx.x, j = threadIdx.x;
  if (j < 16) hs[j] = hin[(size_t)node * 16 + j];
  __syncthreads();
  float s = b1[j];
#pragma unroll
  for (int k = 0; k < 16; ++k) s += hs[k] * w1[k * HD + j];
  ts[j] = silu_f(s);
  __syncthreads();
  float o = b2[j];
#pragma unroll 8
  for (int k = 0; k < HD; ++k) o += ts[k] * w2[k * HD + j];
  hout[(size_t)node * HD + j] = o;
}

// ---------------------------------------------------------------------------
// EGCL edge pass: one workgroup per (graph b, row node r); 256 threads = 8 waves.
// ein = [h_r | h_c | rad | ea]; the h_r half (K=0..63) is IDENTICAL for all 256
// edge rows, so h_r @ ew1[0:64,:] + eb1 is folded into the bias once (f32 GEMV)
// and GEMM1 only runs K-chunks for h_c and rad|ea (24 instead of 40 wmma/wave).
// m = silu(silu(ein@ew1+eb1)@ew2+eb2); w = silu(m@cw1+cb1)@cw2+cb2;
// mi[r] = sum_c m; x_out[r] = (x[r] + sum_c diff*w)*(1-REG).
// Dynamic LDS ~154KB (CDNA5: up to 320KB per workgroup; 2 WGs/WGP).
// ---------------------------------------------------------------------------
__global__ __launch_bounds__(256) void egcl_edge_kernel(
    const float* __restrict__ h_in, const float* __restrict__ x_in,
    const float* __restrict__ eattr,
    const float* __restrict__ ew1, const float* __restrict__ eb1,
    const float* __restrict__ ew2, const float* __restrict__ eb2,
    const float* __restrict__ cw1, const float* __restrict__ cb1,
    const float* __restrict__ cw2, const float* __restrict__ cb2,
    float* __restrict__ mi_out, float* __restrict__ x_out) {
  extern __shared__ char smem[];
  _Float16* h16   = (_Float16*)smem;              // [256][64]  h[b] in f16
  _Float16* aux16 = h16   + NND * HD;             // [256][32]  [rad, ea0..3, 0...]; reused later
  _Float16* t16   = aux16 + NND * 32;             // [256][64]  hidden / c1
  _Float16* m16   = t16   + NND * HD;             // [256][64]  edge message m
  _Float16* w1T   = m16   + NND * HD;             // [64][160]  ew1 transposed (only K>=64 used)
  _Float16* w2T   = w1T   + HD * KP;              // [64][64]   ew2 transposed
  _Float16* w3T   = w2T   + HD * HD;              // [64][64]   cw1 transposed
  float* eb1A = (float*)(w3T + HD * HD);          // [64]  eb1 + h_r @ ew1[0:64,:]
  float* eb2A = eb1A + HD;                        // [64]
  float* cb1A = eb2A + HD;                        // [64]
  float* cw2f = cb1A + HD;                        // [64]
  float* xcs  = cw2f + HD;                        // [256][4]
  float* rads = xcs + NND * 4;                    // [256]
  float* xacc = rads + NND;                       // [4]

  const int tid  = threadIdx.x;
  const int lane = tid & 31;
  const int wave = tid >> 5;
  const int b    = blockIdx.x >> 8;
  const int r    = blockIdx.x & (NND - 1);

  const float* hb = h_in + (size_t)b * NND * HD;
  const float* xb = x_in + (size_t)b * NND * 3;
  const float xr0 = xb[r * 3 + 0], xr1 = xb[r * 3 + 1], xr2 = xb[r * 3 + 2];

  // Stage h[b] f32 -> f16, packed: coalesced global_load_b64 + ds_store_b32.
  {
    const float2* hbv = (const float2*)hb;
    v2h* h16v = (v2h*)h16;
    for (int p = tid; p < NND * HD / 2; p += 256) {
      const float2 v = hbv[p];
      h16v[p] = (v2h){(_Float16)v.x, (_Float16)v.y};
    }
  }
  // Per-edge geometry + edge_attr -> last K-chunk of ein (cols 128..159, zero padded).
  {
    const int c = tid;
    const float xc0 = xb[c * 3 + 0], xc1 = xb[c * 3 + 1], xc2 = xb[c * 3 + 2];
    const float d0 = xr0 - xc0, d1 = xr1 - xc1, d2 = xr2 - xc2;
    const float rad = d0 * d0 + d1 * d1 + d2 * d2;
    xcs[c * 4 + 0] = xc0; xcs[c * 4 + 1] = xc1; xcs[c * 4 + 2] = xc2;
    rads[c] = rad;
    const float* ea = eattr + ((size_t)b * NND * NND + (size_t)r * NND + c) * EDIM;
    v2h* a = (v2h*)(aux16 + c * 32);
    a[0] = (v2h){(_Float16)rad, (_Float16)ea[0]};
    a[1] = (v2h){(_Float16)ea[1], (_Float16)ea[2]};
    a[2] = (v2h){(_Float16)ea[3], (_Float16)0.0f};
#pragma unroll
    for (int k = 3; k < 16; ++k) a[k] = (v2h){(_Float16)0.0f, (_Float16)0.0f};
  }
  if (tid < 4) xacc[tid] = 0.0f;
  // Fold the broadcast h_r contribution into the GEMM1 bias (f32 precision).
  if (tid < HD) {
    float s = eb1[tid];
    for (int k = 0; k < HD; ++k) s += hb[(size_t)r * HD + k] * ew1[k * HD + tid];
    eb1A[tid] = s;
    eb2A[tid] = eb2[tid]; cb1A[tid] = cb1[tid]; cw2f[tid] = cw2[tid];
  }
  // Stage weights TRANSPOSED (coalesced global reads, scattered b16 LDS writes).
  // ew1: only K rows 64..159 needed (h_r rows folded into bias above).
  for (int idx = tid; idx < (KP - HD) * HD; idx += 256) {
    const int k = (idx >> 6) + HD, n = idx & 63;
    w1T[n * KP + k] = (k < KIN) ? (_Float16)ew1[(size_t)k * HD + n] : (_Float16)0.0f;
  }
  for (int idx = tid; idx < HD * HD; idx += 256) {
    const int k = idx >> 6, n = idx & 63;
    w2T[n * HD + k] = (_Float16)ew2[idx];
    w3T[n * HD + k] = (_Float16)cw1[idx];
  }
  __syncthreads();

  const v8f zacc = {0.f, 0.f, 0.f, 0.f, 0.f, 0.f, 0.f, 0.f};
  v8f acc[2][4];

  // ---- GEMM1: t1 = silu(ein @ ew1 + eb1')   (K-chunks: h_c x2, rad|ea x1) ----
#pragma unroll
  for (int mi_ = 0; mi_ < 2; ++mi_)
#pragma unroll
    for (int nt = 0; nt < 4; ++nt) acc[mi_][nt] = zacc;
#pragma unroll
  for (int kc = 0; kc < 3; ++kc) {
    v16h bf[4];
#pragma unroll
    for (int nt = 0; nt < 4; ++nt) bf[nt] = frag_bT(w1T, KP, (kc + 2) * 32, nt * 16, lane);
#pragma unroll
    for (int mi_ = 0; mi_ < 2; ++mi_) {
      const int row0 = (wave * 2 + mi_) * 16;
      const v16h af = (kc < 2) ? frag_a(h16, HD, row0, kc * 32, lane)   // h[c]
                               : frag_a(aux16, 32, row0, 0, lane);      // rad|ea|0
#pragma unroll
      for (int nt = 0; nt < 4; ++nt) acc[mi_][nt] = wmma16(af, bf[nt], acc[mi_][nt]);
    }
  }
#pragma unroll
  for (int mi_ = 0; mi_ < 2; ++mi_) {
    const int rbase = (wave * 2 + mi_) * 16 + ((lane >> 4) << 3);
#pragma unroll
    for (int nt = 0; nt < 4; ++nt) {
      const int col = nt * 16 + (lane & 15);
#pragma unroll
      for (int i = 0; i < 8; ++i)
        t16[(rbase + i) * HD + col] = (_Float16)silu_f(acc[mi_][nt][i] + eb1A[col]);
    }
  }
  __syncthreads();

  // ---- GEMM2: m = silu(t1 @ ew2 + eb2)   (K=64) ----
#pragma unroll
  for (int mi_ = 0; mi_ < 2; ++mi_)
#pragma unroll
    for (int nt = 0; nt < 4; ++nt) acc[mi_][nt] = zacc;
#pragma unroll
  for (int kc = 0; kc < 2; ++kc) {
    v16h bf[4];
#pragma unroll
    for (int nt = 0; nt < 4; ++nt) bf[nt] = frag_bT(w2T, HD, kc * 32, nt * 16, lane);
#pragma unroll
    for (int mi_ = 0; mi_ < 2; ++mi_) {
      const v16h af = frag_a(t16, HD, (wave * 2 + mi_) * 16, kc * 32, lane);
#pragma unroll
      for (int nt = 0; nt < 4; ++nt) acc[mi_][nt] = wmma16(af, bf[nt], acc[mi_][nt]);
    }
  }
#pragma unroll
  for (int mi_ = 0; mi_ < 2; ++mi_) {
    const int rbase = (wave * 2 + mi_) * 16 + ((lane >> 4) << 3);
#pragma unroll
    for (int nt = 0; nt < 4; ++nt) {
      const int col = nt * 16 + (lane & 15);
#pragma unroll
      for (int i = 0; i < 8; ++i)
        m16[(rbase + i) * HD + col] = (_Float16)silu_f(acc[mi_][nt][i] + eb2A[col]);
    }
  }
  __syncthreads();

  // ---- mi segment-sum (all 256 threads; b128 row loads; aux16 reused as f32 scratch) ----
  float* part = (float*)aux16;  // [32][64] f32 partials (8KB <= aux16's 16KB)
  {
    const int rb = tid >> 3;          // 32 row blocks of 8 rows
    const int cb = (tid & 7) * 8;     // 8 col blocks of 8 cols
    float s[8] = {0.f, 0.f, 0.f, 0.f, 0.f, 0.f, 0.f, 0.f};
#pragma unroll
    for (int rr = 0; rr < 8; ++rr) {
      const v8h mv = *(const v8h*)(m16 + (rb * 8 + rr) * HD + cb);
#pragma unroll
      for (int j = 0; j < 8; ++j) s[j] += (float)mv[j];
    }
#pragma unroll
    for (int j = 0; j < 8; ++j) part[rb * HD + cb + j] = s[j];
  }
  __syncthreads();
  if (tid < HD) {  // waves 0-1 whole => no intra-wave divergence around WMMA below
    float s = 0.0f;
#pragma unroll 8
    for (int rb = 0; rb < 32; ++rb) s += part[rb * HD + tid];
    mi_out[((size_t)b * NND + r) * HD + tid] = s;
  }

  // ---- GEMM3: c1 = silu(m @ cw1 + cb1) -> t16 (reuse) ----
#pragma unroll
  for (int mi_ = 0; mi_ < 2; ++mi_)
#pragma unroll
    for (int nt = 0; nt < 4; ++nt) acc[mi_][nt] = zacc;
#pragma unroll
  for (int kc = 0; kc < 2; ++kc) {
    v16h bf[4];
#pragma unroll
    for (int nt = 0; nt < 4; ++nt) bf[nt] = frag_bT(w3T, HD, kc * 32, nt * 16, lane);
#pragma unroll
    for (int mi_ = 0; mi_ < 2; ++mi_) {
      const v16h af = frag_a(m16, HD, (wave * 2 + mi_) * 16, kc * 32, lane);
#pragma unroll
      for (int nt = 0; nt < 4; ++nt) acc[mi_][nt] = wmma16(af, bf[nt], acc[mi_][nt]);
    }
  }
#pragma unroll
  for (int mi_ = 0; mi_ < 2; ++mi_) {
    const int rbase = (wave * 2 + mi_) * 16 + ((lane >> 4) << 3);
#pragma unroll
    for (int nt = 0; nt < 4; ++nt) {
      const int col = nt * 16 + (lane & 15);
#pragma unroll
      for (int i = 0; i < 8; ++i)
        t16[(rbase + i) * HD + col] = (_Float16)silu_f(acc[mi_][nt][i] + cb1A[col]);
    }
  }
  __syncthreads();

  // ---- w = c1 @ cw2 + cb2 (N=1 GEMV, vectorized); accumulate sum_c diff*w in LDS ----
  {
    const int c = tid;
    float wv = cb2[0];
#pragma unroll
    for (int kk = 0; kk < 8; ++kk) {
      const v8h tv = *(const v8h*)(t16 + c * HD + kk * 8);
#pragma unroll
      for (int j = 0; j < 8; ++j) wv += (float)tv[j] * cw2f[kk * 8 + j];
    }
    const float inv = __builtin_amdgcn_rcpf(sqrtf(rads[c]) + EPSV);
    atomicAdd(&xacc[0], (xr0 - xcs[c * 4 + 0]) * inv * wv);  // ds_add_f32
    atomicAdd(&xacc[1], (xr1 - xcs[c * 4 + 1]) * inv * wv);
    atomicAdd(&xacc[2], (xr2 - xcs[c * 4 + 2]) * inv * wv);
  }
  __syncthreads();
  if (tid < 3) {
    const float xr = (tid == 0) ? xr0 : (tid == 1) ? xr1 : xr2;
    x_out[((size_t)b * NND + r) * 3 + tid] = (xr + xacc[tid]) * (1.0f - REGV);
  }
}

// ---------------------------------------------------------------------------
// EGCL node pass: h_out = h + silu([h|mi] @ nw1 + nb1) @ nw2 + nb2.
// 4 waves / 64 nodes per block; WMMA with K=128 then K=64; transposed weights.
// ---------------------------------------------------------------------------
__global__ __launch_bounds__(128) void egcl_node_kernel(
    const float* __restrict__ h_in, const float* __restrict__ mi,
    const float* __restrict__ nw1, const float* __restrict__ nb1,
    const float* __restrict__ nw2, const float* __restrict__ nb2,
    float* __restrict__ h_out) {
  __shared__ _Float16 a16[64 * 128];     // [64 nodes][128 = h|mi]
  __shared__ _Float16 w1T[HD * 128];     // nw1 transposed [64][128]
  __shared__ _Float16 w2T[HD * HD];      // nw2 transposed [64][64]
  __shared__ _Float16 t16[64 * HD];
  __shared__ float    bias1[HD];
  __shared__ float    bias2[HD];
  const int tid = threadIdx.x, lane = tid & 31, wave = tid >> 5;
  const int node0 = blockIdx.x * 64;

  for (int p = tid; p < 64 * 64; p += 128) {   // 4096 f16 pairs, packed stores
    const int rrow = p >> 6, cp = p & 63;
    const int g = node0 + rrow;
    const float2 v = (cp < 32) ? ((const float2*)(h_in + (size_t)g * HD))[cp]
                               : ((const float2*)(mi + (size_t)g * HD))[cp - 32];
    ((v2h*)a16)[p] = (v2h){(_Float16)v.x, (_Float16)v.y};
  }
  for (int idx = tid; idx < 128 * HD; idx += 128) {
    const int k = idx >> 6, n = idx & 63;
    w1T[n * 128 + k] = (_Float16)nw1[idx];
  }
  for (int idx = tid; idx < HD * HD; idx += 128) {
    const int k = idx >> 6, n = idx & 63;
    w2T[n * HD + k] = (_Float16)nw2[idx];
  }
  if (tid < HD) { bias1[tid] = nb1[tid]; bias2[tid] = nb2[tid]; }
  __syncthreads();

  const v8f zacc = {0.f, 0.f, 0.f, 0.f, 0.f, 0.f, 0.f, 0.f};
  v8f acc[4];
#pragma unroll
  for (int nt = 0; nt < 4; ++nt) acc[nt] = zacc;
  const int row0 = wave * 16;
#pragma unroll
  for (int kc = 0; kc < 4; ++kc) {
    const v16h af = frag_a(a16, 128, row0, kc * 32, lane);
#pragma unroll
    for (int nt = 0; nt < 4; ++nt)
      acc[nt] = wmma16(af, frag_bT(w1T, 128, kc * 32, nt * 16, lane), acc[nt]);
  }
  {
    const int rbase = row0 + ((lane >> 4) << 3);
#pragma unroll
    for (int nt = 0; nt < 4; ++nt) {
      const int col = nt * 16 + (lane & 15);
#pragma unroll
      for (int i = 0; i < 8; ++i)
        t16[(rbase + i) * HD + col] = (_Float16)silu_f(acc[nt][i] + bias1[col]);
    }
  }
  __syncthreads();
#pragma unroll
  for (int nt = 0; nt < 4; ++nt) acc[nt] = zacc;
#pragma unroll
  for (int kc = 0; kc < 2; ++kc) {
    const v16h af = frag_a(t16, HD, row0, kc * 32, lane);
#pragma unroll
    for (int nt = 0; nt < 4; ++nt)
      acc[nt] = wmma16(af, frag_bT(w2T, HD, kc * 32, nt * 16, lane), acc[nt]);
  }
  {
    const int rbase = row0 + ((lane >> 4) << 3);
#pragma unroll
    for (int nt = 0; nt < 4; ++nt) {
      const int col = nt * 16 + (lane & 15);
#pragma unroll
      for (int i = 0; i < 8; ++i) {
        const size_t g = (size_t)(node0 + rbase + i) * HD + col;
        h_out[g] = h_in[g] + acc[nt][i] + bias2[col];
      }
    }
  }
}

// ---------------------------------------------------------------------------
// embedding_out: memory-bound (4MB weight) -> split-K VALU GEMV + finisher.
// ---------------------------------------------------------------------------
__global__ __launch_bounds__(512) void zero512_kernel(float* p) { p[threadIdx.x] = 0.0f; }

__global__ __launch_bounds__(64) void embed_out_p1(
    const float* __restrict__ hfin /*[8,16384]*/, const float* __restrict__ w1 /*[16384,64]*/,
    float* __restrict__ acc1 /*[8,64]*/) {
  __shared__ float hs[256];
  const int row = blockIdx.x >> 6;       // 0..7
  const int k0  = (blockIdx.x & 63) * 256;
  const int j   = threadIdx.x;
  for (int idx = j; idx < 256; idx += 64) hs[idx] = hfin[(size_t)row * 16384 + k0 + idx];
  __syncthreads();
  float s = 0.0f;
  for (int k = 0; k < 256; ++k) s += hs[k] * w1[(size_t)(k0 + k) * HD + j];
  atomicAdd(&acc1[row * HD + j], s);
}

__global__ __launch_bounds__(256) void embed_out_p2(
    const float* __restrict__ acc1, const float* __restrict__ b1,
    const float* __restrict__ w2 /*[64,128]*/, const float* __restrict__ b2,
    float* __restrict__ out /*[8,128]*/) {
  __shared__ float t[512];
  const int tid = threadIdx.x;
  for (int idx = tid; idx < 512; idx += 256) t[idx] = silu_f(acc1[idx] + b1[idx & 63]);
  __syncthreads();
  for (int idx = tid; idx < 1024; idx += 256) {
    const int row = idx >> 7, o = idx & 127;
    float s = b2[o];
#pragma unroll 8
    for (int k = 0; k < HD; ++k) s += t[row * HD + k] * w2[k * 128 + o];
    out[idx] = s;
  }
}

// ---------------------------------------------------------------------------
extern "C" void kernel_launch(void* const* d_in, const int* in_sizes, int n_in,
                              void* d_out, int out_size, void* d_ws, size_t ws_size,
                              hipStream_t stream) {
  (void)in_sizes; (void)n_in; (void)out_size; (void)ws_size;
  const float* h_raw = (const float*)d_in[0];
  const float* x_raw = (const float*)d_in[1];
  const float* eattr = (const float*)d_in[2];
  const float* ei_w1 = (const float*)d_in[3];
  const float* ei_b1 = (const float*)d_in[4];
  const float* ei_w2 = (const float*)d_in[5];
  const float* ei_b2 = (const float*)d_in[6];
  const float* e_w1  = (const float*)d_in[7];
  const float* e_b1  = (const float*)d_in[8];
  const float* e_w2  = (const float*)d_in[9];
  const float* e_b2  = (const float*)d_in[10];
  const float* n_w1  = (const float*)d_in[11];
  const float* n_b1  = (const float*)d_in[12];
  const float* n_w2  = (const float*)d_in[13];
  const float* n_b2  = (const float*)d_in[14];
  const float* c_w1  = (const float*)d_in[15];
  const float* c_b1  = (const float*)d_in[16];
  const float* c_w2  = (const float*)d_in[17];
  const float* c_b2  = (const float*)d_in[18];
  const float* eo_w1 = (const float*)d_in[19];
  const float* eo_b1 = (const float*)d_in[20];
  const float* eo_w2 = (const float*)d_in[21];
  const float* eo_b2 = (const float*)d_in[22];
  float* out = (float*)d_out;

  float* ws   = (float*)d_ws;
  float* h0   = ws;
  float* h1   = h0 + (size_t)BATCH * NND * HD;
  float* mi   = h1 + (size_t)BATCH * NND * HD;
  float* x0   = mi + (size_t)BATCH * NND * HD;
  float* x1   = x0 + (size_t)BATCH * NND * 3;
  float* acc1 = x1 + (size_t)BATCH * NND * 3;   // [8*64]

  embed_in_kernel<<<BATCH * NND, 64, 0, stream>>>(h_raw, ei_w1, ei_b1, ei_w2, ei_b2, h0);

  const size_t SMEM_EDGE =
      (size_t)(NND * HD + NND * 32 + NND * HD + NND * HD + HD * KP + 2 * HD * HD) *
          sizeof(_Float16) +
      (size_t)(4 * HD + NND * 4 + NND + 4) * sizeof(float);   // ~154 KB (<= 320KB/WGP)

  const float* xcur = x_raw;
  float* hcur = h0; float* hnxt = h1;
  for (int l = 0; l < LAYERS; ++l) {
    float* xnxt = (l & 1) ? x0 : x1;
    egcl_edge_kernel<<<BATCH * NND, 256, SMEM_EDGE, stream>>>(
        hcur, xcur, eattr,
        e_w1 + (size_t)l * KIN * HD, e_b1 + l * HD,
        e_w2 + (size_t)l * HD * HD,  e_b2 + l * HD,
        c_w1 + (size_t)l * HD * HD,  c_b1 + l * HD,
        c_w2 + (size_t)l * HD,       c_b2 + l,
        mi, xnxt);
    egcl_node_kernel<<<(BATCH * NND) / 64, 128, 0, stream>>>(
        hcur, mi,
        n_w1 + (size_t)l * 2 * HD * HD, n_b1 + l * HD,
        n_w2 + (size_t)l * HD * HD,     n_b2 + l * HD,
        hnxt);
    xcur = xnxt;
    float* t = hcur; hcur = hnxt; hnxt = t;
  }

  zero512_kernel<<<1, 512, 0, stream>>>(acc1);
  embed_out_p1<<<BATCH * 64, 64, 0, stream>>>(hcur, eo_w1, acc1);
  embed_out_p2<<<1, 256, 0, stream>>>(acc1, eo_b1, eo_w2, eo_b2, out);
}
